// MessagePassingComplexUnitaryGCN_53807350284469
// MI455X (gfx1250) — matrix operand order repaired and checked by reading, not compiled, not run on hardware
//
#include <hip/hip_runtime.h>
#include <hip/hip_bf16.h>
#include <math.h>

// ---------------------------------------------------------------------------
// MessagePassingComplexUnitaryGCN on MI455X (gfx1250, wave32, WMMA)
//
// Roofline: ~30 GFLOP of GEMM vs ~5 GB of edge gather/scatter traffic ->
// memory/atomic bound. Strategy:
//   - all GEMMs via v_wmma_f32_16x16x32_bf16 (bf16 activations, f32 accum)
//   - fuse each layer's two matmuls (W-linear -> cReLU -> M-linear) via LDS
//   - bf16 messages (halves gather traffic), f32 native atomics for scatter
//   - edge working set (m + aggr <= 153 MB) fits the 192 MB L2 -> default TH
//   - sorted `batch` -> run-length pooled segment sums (few atomics)
// ---------------------------------------------------------------------------

typedef __attribute__((ext_vector_type(16))) __bf16 v16bf;
typedef __attribute__((ext_vector_type(8)))  __bf16 v8bf;
typedef __attribute__((ext_vector_type(4)))  __bf16 v4bf;
typedef __attribute__((ext_vector_type(2)))  __bf16 v2bf;
typedef __attribute__((ext_vector_type(8)))  float  v8f;

union FragBF { v16bf v; v8bf h[2]; };

__device__ __forceinline__ v8f wmma_bf16(v16bf a, v16bf b, v8f c) {
    // D = A(16x32) x B(32x16) + C, f32 accumulate
    return __builtin_amdgcn_wmma_f32_16x16x32_bf16(
        /*neg_a=*/false, a, /*neg_b=*/false, b,
        /*c_mod=*/(short)0, c, /*reuse_a=*/false, /*reuse_b=*/false);
}

// A-matrix 16x32 bf16 fragment. base points at row 0 of the 16-row tile.
// Lane L: row = L&15; v0-v3 hold K = kb+8*(L>>4)..+7, v4-v7 hold K = +16.
__device__ __forceinline__ v16bf load_a(const __bf16* base, int ld, int kb, int lane) {
    int row = lane & 15;
    int kh  = (lane >> 4) << 3;              // 0 or 8
    const __bf16* p = base + (size_t)row * ld + kb + kh;
    FragBF f;
    f.h[0] = *reinterpret_cast<const v8bf*>(p);        // K = kb+kh .. +7
    f.h[1] = *reinterpret_cast<const v8bf*>(p + 16);   // K = kb+16+kh .. +7
    return f.v;
}

// B-matrix 32x16 bf16 fragment from row-major weights W[O][K] (B[k][o]=W[o][k]).
// Lane L: col o = oBase+(L&15); v0-v3 hold K = kb+8*(L>>4)..+7, v4-v7 hold K = +16.
__device__ __forceinline__ v16bf load_b(const __bf16* w, int ld, int oBase, int kb, int lane) {
    int o  = oBase + (lane & 15);
    int kq = (lane >> 4) << 3;               // 0 or 8
    const __bf16* p = w + (size_t)o * ld + kb + kq;
    FragBF f;
    f.h[0] = *reinterpret_cast<const v8bf*>(p);
    f.h[1] = *reinterpret_cast<const v8bf*>(p + 16);
    return f.v;
}

// ---------------------------------------------------------------------------
// prep kernels
// ---------------------------------------------------------------------------
__global__ void convert_kernel(const float* __restrict__ src, __bf16* __restrict__ dst,
                               __bf16* __restrict__ dstneg, int n) {
    int i = blockIdx.x * blockDim.x + threadIdx.x;
    if (i < n) {
        float v = src[i];
        dst[i] = (__bf16)v;
        if (dstneg) dstneg[i] = (__bf16)(-v);
    }
}

__global__ void factors_kernel(const float* t0r, const float* t0i,
                               const float* t1r, const float* t1i, float* f) {
    if (threadIdx.x == 0 && blockIdx.x == 0) {
        // exp(-i*(tr + i*ti)) = exp(ti) * (cos tr - i sin tr)
        float tr = t0r[0], ti = t0i[0];
        f[0] =  expf(ti) * cosf(tr);           f[1] = -expf(ti) * sinf(tr);            // tf0
        f[2] =  expf(0.5f*ti) * cosf(0.5f*tr); f[3] = -expf(0.5f*ti) * sinf(0.5f*tr);  // ef0
        tr = t1r[0]; ti = t1i[0];
        f[4] =  expf(ti) * cosf(tr);           f[5] = -expf(ti) * sinf(tr);            // tf1
        f[6] =  expf(0.5f*ti) * cosf(0.5f*tr); f[7] = -expf(0.5f*ti) * sinf(0.5f*tr);  // ef1
    }
}

// ---------------------------------------------------------------------------
// Fused layer: per block = 16 nodes, per wave = 16 output cols.
//   phase1: h = crelu(x @ W^T + b)   (complex; xi==nullptr => real input)
//   phase2: m = (h @ M^T + c) * tf   (h staged in LDS, reloaded as A-frags)
// KIN, DH multiples of 32; gridDim.x = N/16; blockDim.x = 2*DH (DH/16 waves).
// ---------------------------------------------------------------------------
template<int KIN, int DH>
__global__ void __launch_bounds__(DH * 2)
fused_layer_kernel(const __bf16* __restrict__ xr, const __bf16* __restrict__ xi,
                   const __bf16* __restrict__ Wr, const __bf16* __restrict__ Wi,
                   const __bf16* __restrict__ Win,
                   const float* __restrict__ br, const float* __restrict__ bi,
                   const __bf16* __restrict__ Mr, const __bf16* __restrict__ Mi,
                   const __bf16* __restrict__ Min,
                   const float* __restrict__ cr, const float* __restrict__ ci,
                   const float* __restrict__ fac,   // fac[0]=tfr fac[1]=tfi
                   __bf16* __restrict__ hr_out, __bf16* __restrict__ hi_out,
                   __bf16* __restrict__ mr_out, __bf16* __restrict__ mi_out)
{
    __shared__ __align__(16) __bf16 hrS[16 * DH];
    __shared__ __align__(16) __bf16 hiS[16 * DH];

    const int lane  = threadIdx.x & 31;
    const int wave  = threadIdx.x >> 5;
    const int cBase = wave * 16;
    const int rBase = blockIdx.x * 16;
    const __bf16* Ar = xr + (size_t)rBase * KIN;

    v8f accR = {0,0,0,0,0,0,0,0};
    v8f accI = {0,0,0,0,0,0,0,0};

    if (xi == nullptr) {   // layer-0: real input, hr = x@Wr, hi = x@Wi
        for (int kb = 0; kb < KIN; kb += 32) {
            v16bf a   = load_a(Ar, KIN, kb, lane);
            v16bf brf = load_b(Wr, KIN, cBase, kb, lane);
            v16bf bif = load_b(Wi, KIN, cBase, kb, lane);
            accR = wmma_bf16(a, brf, accR);
            accI = wmma_bf16(a, bif, accI);
        }
    } else {               // complex: hr = xr@Wr + xi@(-Wi), hi = xr@Wi + xi@Wr
        const __bf16* Ai = xi + (size_t)rBase * KIN;
        for (int kb = 0; kb < KIN; kb += 32) {
            v16bf ar  = load_a(Ar, KIN, kb, lane);
            v16bf ai  = load_a(Ai, KIN, kb, lane);
            v16bf brf = load_b(Wr,  KIN, cBase, kb, lane);
            v16bf bif = load_b(Wi,  KIN, cBase, kb, lane);
            v16bf bnf = load_b(Win, KIN, cBase, kb, lane);
            accR = wmma_bf16(ar, brf, accR);
            accR = wmma_bf16(ai, bnf, accR);
            accI = wmma_bf16(ar, bif, accI);
            accI = wmma_bf16(ai, brf, accI);
        }
    }

    // bias + complex ReLU; write h to global and to LDS for phase 2.
    const int   col = cBase + (lane & 15);
    const float bre = br[col], bim = bi[col];
#pragma unroll
    for (int j = 0; j < 8; ++j) {
        int row = j + ((lane >> 4) << 3);   // C/D layout: vgpr j -> M=j / j+8
        float hr = fmaxf(accR[j] + bre, 0.0f);
        float hi = fmaxf(accI[j] + bim, 0.0f);
        __bf16 hrb = (__bf16)hr, hib = (__bf16)hi;
        hrS[row * DH + col] = hrb;
        hiS[row * DH + col] = hib;
        size_t g = (size_t)(rBase + row) * DH + col;
        hr_out[g] = hrb;
        hi_out[g] = hib;
    }
    __syncthreads();

    // phase 2: m = (h @ M^T + c) * tf, K = DH (square message matrix)
    v8f mR = {0,0,0,0,0,0,0,0};
    v8f mI = {0,0,0,0,0,0,0,0};
    for (int kb = 0; kb < DH; kb += 32) {
        v16bf ar  = load_a(hrS, DH, kb, lane);
        v16bf ai  = load_a(hiS, DH, kb, lane);
        v16bf brf = load_b(Mr,  DH, cBase, kb, lane);
        v16bf bif = load_b(Mi,  DH, cBase, kb, lane);
        v16bf bnf = load_b(Min, DH, cBase, kb, lane);
        mR = wmma_bf16(ar, brf, mR);
        mR = wmma_bf16(ai, bnf, mR);
        mI = wmma_bf16(ar, bif, mI);
        mI = wmma_bf16(ai, brf, mI);
    }
    const float tfr = fac[0], tfi = fac[1];
    const float cre = cr[col], cim = ci[col];
#pragma unroll
    for (int j = 0; j < 8; ++j) {
        int row = j + ((lane >> 4) << 3);
        float pr = mR[j] + cre;
        float pi = mI[j] + cim;
        size_t g = (size_t)(rBase + row) * DH + col;
        mr_out[g] = (__bf16)(pr * tfr - pi * tfi);
        mi_out[g] = (__bf16)(pr * tfi + pi * tfr);
    }
}

// ---------------------------------------------------------------------------
// Edge scatter: aggr[dst] += m[src]. Each thread owns 4 consecutive features
// (one 8-byte bf16x4 load per matrix, 4 f32 atomics each). Atomics to
// consecutive addresses; the m/aggr working set is L2-resident.
// ---------------------------------------------------------------------------
template<int DH>
__global__ void edge_scatter_kernel(const long long* __restrict__ src,
                                    const long long* __restrict__ dst,
                                    const __bf16* __restrict__ mr,
                                    const __bf16* __restrict__ mi,
                                    float* __restrict__ ar, float* __restrict__ ai,
                                    int nE)
{
    constexpr int TPE = DH / 4;              // threads per edge
    long long t = (long long)blockIdx.x * blockDim.x + threadIdx.x;
    int e = (int)(t / TPE);
    int j = ((int)(t % TPE)) * 4;
    if (e < nE) {
        long long s = src[e], d = dst[e];
        v4bf vr = *reinterpret_cast<const v4bf*>(mr + (size_t)s * DH + j);
        v4bf vi = *reinterpret_cast<const v4bf*>(mi + (size_t)s * DH + j);
        float* pr = ar + (size_t)d * DH + j;
        float* pi = ai + (size_t)d * DH + j;
#pragma unroll
        for (int k = 0; k < 4; ++k) {
            atomicAdd(pr + k, (float)vr[k]);
            atomicAdd(pi + k, (float)vi[k]);
        }
    }
}

// ---------------------------------------------------------------------------
// combine: z = h*ef + aggr*(1-ef)   (elementwise complex, 2 features/thread)
// ---------------------------------------------------------------------------
__global__ void combine_kernel(const __bf16* __restrict__ hr, const __bf16* __restrict__ hi,
                               const float* __restrict__ ar, const float* __restrict__ ai,
                               const float* __restrict__ fac,   // ef
                               __bf16* __restrict__ outr, __bf16* __restrict__ outi, int n2)
{
    int idx = blockIdx.x * blockDim.x + threadIdx.x;   // pair index
    if (idx >= n2) return;
    float efr = fac[0], efi = fac[1];
    float omr = 1.0f - efr, omi = -efi;
    v2bf xrv = *reinterpret_cast<const v2bf*>(hr + 2 * (size_t)idx);
    v2bf xiv = *reinterpret_cast<const v2bf*>(hi + 2 * (size_t)idx);
    v2bf zr, zi;
#pragma unroll
    for (int k = 0; k < 2; ++k) {
        float xr = (float)xrv[k], xi = (float)xiv[k];
        float gr = ar[2 * (size_t)idx + k], gi = ai[2 * (size_t)idx + k];
        zr[k] = (__bf16)(xr * efr - xi * efi + gr * omr - gi * omi);
        zi[k] = (__bf16)(xr * efi + xi * efr + gr * omi + gi * omr);
    }
    *reinterpret_cast<v2bf*>(outr + 2 * (size_t)idx) = zr;
    *reinterpret_cast<v2bf*>(outi + 2 * (size_t)idx) = zi;
}

// ---------------------------------------------------------------------------
// Final combine + |z| + sorted-segment-sum pooling (batch is sorted, so each
// thread keeps a running sum per segment and flushes on segment change —
// ~1-2 atomics per block instead of one per node-feature).
// blockDim = (64, 4), block handles 128 nodes.
// ---------------------------------------------------------------------------
__global__ void combine_pool_kernel(const __bf16* __restrict__ hr, const __bf16* __restrict__ hi,
                                    const float* __restrict__ ar, const float* __restrict__ ai,
                                    const float* __restrict__ fac,
                                    const long long* __restrict__ batch,
                                    float* __restrict__ sums, float* __restrict__ counts,
                                    int nNodes)
{
    const int j     = threadIdx.x;           // feature 0..63
    const int row   = threadIdx.y;           // 0..3
    const int nBase = blockIdx.x * 128;
    const float efr = fac[0], efi = fac[1];
    const float omr = 1.0f - efr, omi = -efi;

    float acc = 0.0f, cnt = 0.0f;
    int curg = -1;
    int nEnd = nBase + 128; if (nEnd > nNodes) nEnd = nNodes;
    for (int n = nBase + row; n < nEnd; n += 4) {
        int g = (int)batch[n];
        if (g != curg) {
            if (curg >= 0) {
                atomicAdd(&sums[curg * 64 + j], acc);
                if (j == 0) atomicAdd(&counts[curg], cnt);
            }
            acc = 0.0f; cnt = 0.0f; curg = g;
        }
        size_t o = (size_t)n * 64 + j;
        float xr = (float)hr[o], xi = (float)hi[o];
        float gr = ar[o], gi = ai[o];
        float zr = xr * efr - xi * efi + gr * omr - gi * omi;
        float zi = xr * efi + xi * efr + gr * omi + gi * omr;
        acc += sqrtf(zr * zr + zi * zi);
        cnt += 1.0f;
    }
    if (curg >= 0) {
        atomicAdd(&sums[curg * 64 + j], acc);
        if (j == 0) atomicAdd(&counts[curg], cnt);
    }
}

// mean + log_softmax over 64 classes; one wave (32 lanes x 2 cols) per graph.
__global__ void pool_softmax_kernel(const float* __restrict__ sums,
                                    const float* __restrict__ counts,
                                    float* __restrict__ out)
{
    int g = blockIdx.x, l = threadIdx.x;   // 32 lanes
    float c  = fmaxf(counts[g], 1.0f);
    float v0 = sums[g * 64 + l]      / c;
    float v1 = sums[g * 64 + l + 32] / c;
    float m = fmaxf(v0, v1);
    for (int off = 16; off > 0; off >>= 1) m = fmaxf(m, __shfl_xor(m, off, 32));
    float s = expf(v0 - m) + expf(v1 - m);
    for (int off = 16; off > 0; off >>= 1) s += __shfl_xor(s, off, 32);
    float lse = m + logf(s);
    out[g * 64 + l]      = v0 - lse;
    out[g * 64 + l + 32] = v1 - lse;
}

// ---------------------------------------------------------------------------
extern "C" void kernel_launch(void* const* d_in, const int* in_sizes, int n_in,
                              void* d_out, int out_size, void* d_ws, size_t ws_size,
                              hipStream_t stream)
{
    (void)in_sizes; (void)n_in; (void)out_size; (void)ws_size;
    constexpr int N = 100000, E = 1600000, G = 64;
    constexpr int DIN = 128, DHH = 128, DOUT = 64;

    const float* x    = (const float*)d_in[0];
    const float* W0r  = (const float*)d_in[1];
    const float* W0i  = (const float*)d_in[2];
    const float* b0r  = (const float*)d_in[3];
    const float* b0i  = (const float*)d_in[4];
    const float* M0r  = (const float*)d_in[5];
    const float* M0i  = (const float*)d_in[6];
    const float* c0r  = (const float*)d_in[7];
    const float* c0i  = (const float*)d_in[8];
    const float* t0r  = (const float*)d_in[9];
    const float* t0i  = (const float*)d_in[10];
    const float* W1r  = (const float*)d_in[11];
    const float* W1i  = (const float*)d_in[12];
    const float* b1r  = (const float*)d_in[13];
    const float* b1i  = (const float*)d_in[14];
    const float* M1r  = (const float*)d_in[15];
    const float* M1i  = (const float*)d_in[16];
    const float* c1r  = (const float*)d_in[17];
    const float* c1i  = (const float*)d_in[18];
    const float* t1r  = (const float*)d_in[19];
    const float* t1i  = (const float*)d_in[20];
    const long long* edge  = (const long long*)d_in[21];  // int64 per reference
    const long long* batch = (const long long*)d_in[22];
    const long long* esrc = edge;
    const long long* edst = edge + E;

    // ---- workspace layout (256B aligned), regions reused across lifetimes --
    char* ws = (char*)d_ws;
    size_t off = 0;
    auto alloc = [&](size_t bytes) -> char* {
        off = (off + 255) & ~(size_t)255;
        char* p = ws + off; off += bytes; return p;
    };
    float*  fac   = (float*) alloc(64);                      // tf0,ef0,tf1,ef1
    __bf16* wW0r  = (__bf16*)alloc((size_t)DHH*DIN*2);
    __bf16* wW0i  = (__bf16*)alloc((size_t)DHH*DIN*2);
    __bf16* wW0n  = (__bf16*)alloc((size_t)DHH*DIN*2);
    __bf16* wM0r  = (__bf16*)alloc((size_t)DHH*DHH*2);
    __bf16* wM0i  = (__bf16*)alloc((size_t)DHH*DHH*2);
    __bf16* wM0n  = (__bf16*)alloc((size_t)DHH*DHH*2);
    __bf16* wW1r  = (__bf16*)alloc((size_t)DOUT*DHH*2);
    __bf16* wW1i  = (__bf16*)alloc((size_t)DOUT*DHH*2);
    __bf16* wW1n  = (__bf16*)alloc((size_t)DOUT*DHH*2);
    __bf16* wM1r  = (__bf16*)alloc((size_t)DOUT*DOUT*2);
    __bf16* wM1i  = (__bf16*)alloc((size_t)DOUT*DOUT*2);
    __bf16* wM1n  = (__bf16*)alloc((size_t)DOUT*DOUT*2);
    __bf16* xb    = (__bf16*)alloc((size_t)N*DIN*2);         // x in bf16
    __bf16* h0r   = (__bf16*)alloc((size_t)N*DHH*2);         // h0 / x1 (in-place)
    __bf16* h0i   = (__bf16*)alloc((size_t)N*DHH*2);
    __bf16* msg0r = (__bf16*)alloc((size_t)N*DHH*2);         // m0; reused as h1/m1
    __bf16* msg0i = (__bf16*)alloc((size_t)N*DHH*2);
    float*  agg0r = (float*) alloc((size_t)N*DHH*4);         // aggr0; reused as aggr1
    float*  agg0i = (float*) alloc((size_t)N*DHH*4);
    float*  sums  = (float*) alloc((size_t)G*DOUT*4);
    float*  cnts  = (float*) alloc((size_t)G*4);
    // layer-1 aliases (lifetimes disjoint from m0 / aggr0)
    __bf16* h1r   = msg0r;
    __bf16* h1i   = msg0r + (size_t)N*DOUT;
    __bf16* msg1r = msg0i;
    __bf16* msg1i = msg0i + (size_t)N*DOUT;
    float*  agg1r = agg0r;
    float*  agg1i = agg0i;

    auto cvt = [&](const float* s, __bf16* d, __bf16* dn, int n) {
        convert_kernel<<<(n + 255) / 256, 256, 0, stream>>>(s, d, dn, n);
    };

    // ---- prep ----
    cvt(x,   xb,   nullptr, N * DIN);
    cvt(W0r, wW0r, nullptr, DHH * DIN);
    cvt(W0i, wW0i, wW0n,    DHH * DIN);
    cvt(M0r, wM0r, nullptr, DHH * DHH);
    cvt(M0i, wM0i, wM0n,    DHH * DHH);
    cvt(W1r, wW1r, nullptr, DOUT * DHH);
    cvt(W1i, wW1i, wW1n,    DOUT * DHH);
    cvt(M1r, wM1r, nullptr, DOUT * DOUT);
    cvt(M1i, wM1i, wM1n,    DOUT * DOUT);
    factors_kernel<<<1, 32, 0, stream>>>(t0r, t0i, t1r, t1i, fac);

    // ---- layer 0 ----
    hipMemsetAsync(agg0r, 0, (size_t)N * DHH * 4, stream);
    hipMemsetAsync(agg0i, 0, (size_t)N * DHH * 4, stream);
    fused_layer_kernel<DIN, DHH><<<N / 16, DHH * 2, 0, stream>>>(
        xb, nullptr, wW0r, wW0i, wW0n, b0r, b0i,
        wM0r, wM0i, wM0n, c0r, c0i, fac + 0,
        h0r, h0i, msg0r, msg0i);
    {
        long long thr0 = (long long)E * (DHH / 4);            // 32 threads/edge
        edge_scatter_kernel<DHH><<<(unsigned)((thr0 + 255) / 256), 256, 0, stream>>>(
            esrc, edst, msg0r, msg0i, agg0r, agg0i, E);
    }
    combine_kernel<<<(N * DHH / 2 + 255) / 256, 256, 0, stream>>>(
        h0r, h0i, agg0r, agg0i, fac + 2, h0r, h0i, N * DHH / 2);  // in-place -> x1

    // ---- layer 1 ----
    hipMemsetAsync(agg1r, 0, (size_t)N * DOUT * 4, stream);
    hipMemsetAsync(agg1i, 0, (size_t)N * DOUT * 4, stream);
    fused_layer_kernel<DHH, DOUT><<<N / 16, DOUT * 2, 0, stream>>>(
        h0r, h0i, wW1r, wW1i, wW1n, b1r, b1i,
        wM1r, wM1i, wM1n, c1r, c1i, fac + 4,
        h1r, h1i, msg1r, msg1i);
    {
        long long thr1 = (long long)E * (DOUT / 4);           // 16 threads/edge
        edge_scatter_kernel<DOUT><<<(unsigned)((thr1 + 255) / 256), 256, 0, stream>>>(
            esrc, edst, msg1r, msg1i, agg1r, agg1i, E);
    }

    // ---- pooling + log_softmax ----
    hipMemsetAsync(sums, 0, (size_t)G * DOUT * 4, stream);
    hipMemsetAsync(cnts, 0, (size_t)G * 4, stream);
    combine_pool_kernel<<<(N + 127) / 128, dim3(64, 4), 0, stream>>>(
        h1r, h1i, agg1r, agg1i, fac + 6, batch, sums, cnts, N);
    pool_softmax_kernel<<<G, 32, 0, stream>>>(sums, cnts, (float*)d_out);
}
